// GCNLayer_89043261980850
// MI455X (gfx1250) — matrix operand order repaired and checked by reading, not compiled, not run on hardware
//
#include <hip/hip_runtime.h>
#include <stdint.h>

typedef __bf16 bf16;
typedef __bf16 v16bf __attribute__((ext_vector_type(16)));
typedef float  v8f   __attribute__((ext_vector_type(8)));
typedef float  v8fv  __attribute__((ext_vector_type(8)));
typedef __bf16 v8bfv __attribute__((ext_vector_type(8)));

#define LEAKY_SLOPE 0.01f
#define LN_EPS      1e-5f

// ---------- CDNA5 async global->LDS copy (guarded; falls back to sync copy) ----
#if defined(__has_builtin)
#if __has_builtin(__builtin_amdgcn_global_load_async_to_lds_b128)
#define HAVE_ASYNC_LDS 1
#endif
#endif

#if defined(HAVE_ASYNC_LDS)
// Builtin signature (from hipcc diagnostic): param0 = AS(1) ptr to gcc-vector int4.
typedef int gv4i __attribute__((__vector_size__(16)));
typedef __attribute__((address_space(1))) gv4i* as1_v4i_ptr;
typedef __attribute__((address_space(3))) gv4i* as3_v4i_ptr;

__device__ __forceinline__ void async_copy_b128(const void* g, void* l) {
  __builtin_amdgcn_global_load_async_to_lds_b128(
      (as1_v4i_ptr)((void*)g), (as3_v4i_ptr)l, 0, 0);
}
__device__ __forceinline__ void async_wait0() {
#if __has_builtin(__builtin_amdgcn_s_wait_asynccnt)
  __builtin_amdgcn_s_wait_asynccnt(0);
#else
  asm volatile("s_wait_asynccnt 0" ::: "memory");
#endif
}
#endif

// ---------- helpers ----------
// Convert 16 fp32 (two float8 register blocks) -> 16 bf16, store 32B to LDS.
__device__ __forceinline__ void cvt_store16(void* dst, v8fv f0, v8fv f1) {
  union { v8bfv v; uint4 q; } a, b;
  a.v = __builtin_convertvector(f0, v8bfv);   // -> v_cvt_pk_bf16_f32 x4
  b.v = __builtin_convertvector(f1, v8bfv);
  ((uint4*)dst)[0] = a.q;
  ((uint4*)dst)[1] = b.q;
}

union FragA { uint4 q[2]; v16bf v; };

// Load a 32B B-fragment (col = 16n + lr, K = 8h..8h+7 and 16+8h..16+8h+7) from an 80B-pitch tile.
__device__ __forceinline__ void load_bfrag(FragA& f, const char* base, int n, int lr, int hh) {
  const char* p = base + (16 * n + lr) * 80 + 16 * hh;
  f.q[0] = ((const uint4*)p)[0];
  f.q[1] = ((const uint4*)(p + 32))[0];
}

// ---------------------------------------------------------------------------
// k1: deg = row-sum(adj);  dis = deg>0 ? rsqrt(deg) : 0.   grid (4096,8) x 256
// ---------------------------------------------------------------------------
__global__ __launch_bounds__(256) void k_deg(const float* __restrict__ adj,
                                             float* __restrict__ dis) {
  __shared__ float red[256];
  const int n = blockIdx.x, b = blockIdx.y;
  const size_t base = ((size_t)b * 4096 + n) * 4096;
  const float4* a4 = (const float4*)(adj + base);
  float s = 0.f;
  for (int i = threadIdx.x; i < 1024; i += 256) {
    const float4 v = a4[i];
    s += (v.x + v.y) + (v.z + v.w);
  }
  red[threadIdx.x] = s;
  __syncthreads();
  for (int off = 128; off > 0; off >>= 1) {
    if (threadIdx.x < off) red[threadIdx.x] += red[threadIdx.x + off];
    __syncthreads();
  }
  if (threadIdx.x == 0) {
    const float d = red[0];
    dis[(size_t)b * 4096 + n] = (d > 0.f) ? rsqrtf(d) : 0.f;
  }
}

// ---------------------------------------------------------------------------
// k2: X' = dis[m] * LeakyReLU(feats @ W^T + bias), bf16, transposed Xt[b][o][m]
// grid (32,8) x 256 (8 waves: 16 rows x 128 cols each), K=128 in 4 steps
// ---------------------------------------------------------------------------
__global__ __launch_bounds__(256) void k_feat(const float* __restrict__ feats,
                                              const float* __restrict__ W,
                                              const float* __restrict__ bias,
                                              const float* __restrict__ dis,
                                              uint16_t* __restrict__ Xt) {
  // 80-byte pitch (40 bf16): 20-bank stride -> conflict-free b128 frag loads
  __shared__ __align__(16) bf16 Abf[128 * 40];
  __shared__ __align__(16) bf16 Bbf[128 * 40];
  __shared__ float disL[128], biasL[128];

  const int b  = blockIdx.y;
  const int m0 = blockIdx.x * 128;
  const int tid = threadIdx.x;
  const int w = tid >> 5, lane = tid & 31, lr = lane & 15, hh = lane >> 4;

  if (tid < 128) {
    disL[tid]  = dis[(size_t)b * 4096 + m0 + tid];
    biasL[tid] = bias[tid];
  }

  v8f acc[8];
#pragma unroll
  for (int n = 0; n < 8; ++n)
#pragma unroll
    for (int j = 0; j < 8; ++j) acc[n][j] = 0.f;

  const int srow = tid >> 1, sh = tid & 1;
  const float* aSrc = feats + ((size_t)(b * 4096 + m0 + srow)) * 128 + 16 * sh;
  const float* bSrc = W + (size_t)srow * 128 + 16 * sh;
  char* aDst = (char*)Abf + srow * 80 + 32 * sh;
  char* bDst = (char*)Bbf + srow * 80 + 32 * sh;

  for (int k0 = 0; k0 < 128; k0 += 32) {
    __syncthreads();
    {
      const v8fv f0 = *(const v8fv*)(aSrc + k0);
      const v8fv f1 = *(const v8fv*)(aSrc + k0 + 8);
      const v8fv g0 = *(const v8fv*)(bSrc + k0);
      const v8fv g1 = *(const v8fv*)(bSrc + k0 + 8);
      cvt_store16(aDst, f0, f1);
      cvt_store16(bDst, g0, g1);
    }
    __syncthreads();

    FragA ua;
    const char* abase = (const char*)Abf + (w * 16 + lr) * 80;
    ua.q[0] = *(const uint4*)(abase + 16 * hh);
    ua.q[1] = *(const uint4*)(abase + 32 + 16 * hh);

    // ping-pong B fragments so wmma(n) overlaps ds_load(n+1)
    FragA ub[2];
    load_bfrag(ub[0], (const char*)Bbf, 0, lr, hh);
#pragma unroll
    for (int n = 0; n < 8; ++n) {
      if (n < 7) load_bfrag(ub[(n + 1) & 1], (const char*)Bbf, n + 1, lr, hh);
      acc[n] = __builtin_amdgcn_wmma_f32_16x16x32_bf16(false, ua.v, false, ub[n & 1].v,
                                                       (short)0, acc[n], false, false);
    }
  }

#pragma unroll
  for (int n = 0; n < 8; ++n) {
    const int o = 16 * n + lr;
#pragma unroll
    for (int j = 0; j < 8; ++j) {
      const int r = w * 16 + j + 8 * hh;
      float x = acc[n][j] + biasL[o];
      x = (x >= 0.f) ? x : LEAKY_SLOPE * x;
      x *= disL[r];
      const bf16 hx = (bf16)x;
      Xt[((size_t)(b * 128 + o)) * 4096 + m0 + r] = __builtin_bit_cast(uint16_t, hx);
    }
  }
}

// ---------------------------------------------------------------------------
// k3: Z = dis[n] .* (adj_bf16 @ X')  -> d_out (pre-LayerNorm)
// grid (32,8) x 256.  Double-buffered LDS pipeline, one barrier per K-step:
//   iter k: issue A(k+32) global loads + B(k+32) async->LDS[nxt],
//           8x WMMA from LDS[cur], convert/store A regs -> LDS[nxt].
// ---------------------------------------------------------------------------
__global__ __launch_bounds__(256) void k_bmm(const float* __restrict__ adj,
                                             const uint16_t* __restrict__ Xt,
                                             const float* __restrict__ dis,
                                             float* __restrict__ out) {
  __shared__ __align__(16) bf16 Abf[2][128 * 40];
  __shared__ __align__(16) bf16 Bbf[2][128 * 40];
  __shared__ float disL[128];

  const int b  = blockIdx.y;
  const int r0 = blockIdx.x * 128;
  const int tid = threadIdx.x;
  const int w = tid >> 5, lane = tid & 31, lr = lane & 15, hh = lane >> 4;

  if (tid < 128) disL[tid] = dis[(size_t)b * 4096 + r0 + tid];

  v8f acc[8];
#pragma unroll
  for (int n = 0; n < 8; ++n)
#pragma unroll
    for (int j = 0; j < 8; ++j) acc[n][j] = 0.f;

  const int srow = tid >> 1, sh = tid & 1;
  const float*    aSrc = adj + ((size_t)b * 4096 + r0 + srow) * 4096 + 16 * sh;
  const uint16_t* xSrc = Xt + ((size_t)(b * 128 + srow)) * 4096 + 16 * sh;
  char* aDst[2] = {(char*)&Abf[0][0] + srow * 80 + 32 * sh,
                   (char*)&Abf[1][0] + srow * 80 + 32 * sh};
  char* bDst[2] = {(char*)&Bbf[0][0] + srow * 80 + 32 * sh,
                   (char*)&Bbf[1][0] + srow * 80 + 32 * sh};

  // ---- prologue: fill buffer 0 with tile k0 = 0 ----
  {
    const v8fv f0 = *(const v8fv*)(aSrc);
    const v8fv f1 = *(const v8fv*)(aSrc + 8);
    cvt_store16(aDst[0], f0, f1);
#if defined(HAVE_ASYNC_LDS)
    async_copy_b128(xSrc, bDst[0]);
    async_copy_b128(xSrc + 8, bDst[0] + 16);
    async_wait0();
#else
    const uint4* bs = (const uint4*)xSrc;
    ((uint4*)bDst[0])[0] = bs[0];
    ((uint4*)bDst[0])[1] = bs[1];
#endif
  }
  __syncthreads();

  for (int k0 = 0; k0 < 4096; k0 += 32) {
    const int cur = (k0 >> 5) & 1;
    const int nxt = cur ^ 1;
    const bool more = (k0 + 32) < 4096;

    v8fv nf0, nf1;
#if !defined(HAVE_ASYNC_LDS)
    uint4 nb0, nb1;
#endif
    if (more) {
      // issue next A tile global loads (fp32) into registers
      nf0 = *(const v8fv*)(aSrc + k0 + 32);
      nf1 = *(const v8fv*)(aSrc + k0 + 40);
#if defined(HAVE_ASYNC_LDS)
      // next B tile: async global -> LDS, no VGPR round trip
      async_copy_b128(xSrc + k0 + 32, bDst[nxt]);
      async_copy_b128(xSrc + k0 + 40, bDst[nxt] + 16);
#else
      nb0 = ((const uint4*)(xSrc + k0 + 32))[0];
      nb1 = ((const uint4*)(xSrc + k0 + 32))[1];
#endif
      __builtin_prefetch(aSrc + k0 + 64, 0, 1);
    }

    // ---- compute from current buffer ----
    FragA ua;
    const char* abase = (const char*)&Abf[cur][0] + (w * 16 + lr) * 80;
    ua.q[0] = *(const uint4*)(abase + 16 * hh);
    ua.q[1] = *(const uint4*)(abase + 32 + 16 * hh);

    FragA ub[2];
    load_bfrag(ub[0], (const char*)&Bbf[cur][0], 0, lr, hh);
#pragma unroll
    for (int n = 0; n < 8; ++n) {
      if (n < 7) load_bfrag(ub[(n + 1) & 1], (const char*)&Bbf[cur][0], n + 1, lr, hh);
      acc[n] = __builtin_amdgcn_wmma_f32_16x16x32_bf16(false, ua.v, false, ub[n & 1].v,
                                                       (short)0, acc[n], false, false);
    }

    if (more) {
      cvt_store16(aDst[nxt], nf0, nf1);   // convert A regs -> next LDS buffer
#if !defined(HAVE_ASYNC_LDS)
      ((uint4*)bDst[nxt])[0] = nb0;
      ((uint4*)bDst[nxt])[1] = nb1;
#endif
    }
#if defined(HAVE_ASYNC_LDS)
    async_wait0();
#endif
    __syncthreads();
  }

  // epilogue: scale by dis[n] (left degree factor), write pre-LN result
#pragma unroll
  for (int n = 0; n < 8; ++n) {
    const int o = 16 * n + lr;
#pragma unroll
    for (int j = 0; j < 8; ++j) {
      const int r = w * 16 + j + 8 * hh;
      out[((size_t)b * 4096 + r0 + r) * 128 + o] = acc[n][j] * disL[r];
    }
  }
}

// ---------------------------------------------------------------------------
// k4: in-place LayerNorm over last dim (128).  One wave32 per row.
// ---------------------------------------------------------------------------
__global__ __launch_bounds__(256) void k_ln(float* __restrict__ out) {
  const int row  = blockIdx.x * 8 + (threadIdx.x >> 5);
  const int lane = threadIdx.x & 31;
  float4* p = (float4*)(out + (size_t)row * 128);
  const float4 v = p[lane];
  float s1 = (v.x + v.y) + (v.z + v.w);
  float s2 = (v.x * v.x + v.y * v.y) + (v.z * v.z + v.w * v.w);
#pragma unroll
  for (int m = 16; m > 0; m >>= 1) {
    s1 += __shfl_xor(s1, m, 32);
    s2 += __shfl_xor(s2, m, 32);
  }
  const float mu = s1 * (1.f / 128.f);
  float var = s2 * (1.f / 128.f) - mu * mu;
  var = var < 0.f ? 0.f : var;
  const float rs = rsqrtf(var + LN_EPS);
  float4 o;
  o.x = (v.x - mu) * rs;
  o.y = (v.y - mu) * rs;
  o.z = (v.z - mu) * rs;
  o.w = (v.w - mu) * rs;
  p[lane] = o;
}

// ---------------------------------------------------------------------------
extern "C" void kernel_launch(void* const* d_in, const int* in_sizes, int n_in,
                              void* d_out, int out_size, void* d_ws, size_t ws_size,
                              hipStream_t stream) {
  const float* feats = (const float*)d_in[0];  // [8,4096,128]
  const float* adj   = (const float*)d_in[1];  // [8,4096,4096]
  const float* W     = (const float*)d_in[2];  // [128,128]
  const float* bias  = (const float*)d_in[3];  // [128]
  float* out = (float*)d_out;                  // [8,4096,128]

  // workspace: dis (128 KB) @ 0 ; Xt bf16 (8 MB) @ 256 KB
  float*    dis = (float*)d_ws;
  uint16_t* Xt  = (uint16_t*)((char*)d_ws + (size_t)(256 << 10));

  k_deg <<<dim3(4096, 8), 256, 0, stream>>>(adj, dis);
  k_feat<<<dim3(32, 8),   256, 0, stream>>>(feats, W, bias, dis, Xt);
  k_bmm <<<dim3(32, 8),   256, 0, stream>>>(adj, Xt, dis, out);
  k_ln  <<<dim3(4096),    256, 0, stream>>>(out);
}
